// SelfAttention_3865470567029
// MI455X (gfx1250) — matrix operand order repaired and checked by reading, not compiled, not run on hardware
//
#include <hip/hip_runtime.h>

typedef _Float16 h16;
typedef __attribute__((ext_vector_type(8)))  _Float16 v8h;
typedef __attribute__((ext_vector_type(16))) _Float16 v16h;
typedef __attribute__((ext_vector_type(8)))  float    v8f;

static constexpr int Bb = 4, Tt = 2048, Cc = 1024, Hh = 16, Dd = 64;

#if __has_builtin(__builtin_amdgcn_exp2f)
__device__ __forceinline__ float fast_exp2(float x) { return __builtin_amdgcn_exp2f(x); }
#else
__device__ __forceinline__ float fast_exp2(float x) { return exp2f(x); }
#endif

__device__ __forceinline__ v16h cat8(v8h lo, v8h hi) {
    return __builtin_shufflevector(lo, hi, 0,1,2,3,4,5,6,7,8,9,10,11,12,13,14,15);
}
__device__ __forceinline__ v16h ldfrag(const h16* p0, const h16* p1) {
    v8h lo = *(const v8h*)p0;
    v8h hi = *(const v8h*)p1;
    return cat8(lo, hi);
}
__device__ __forceinline__ v8f wmma_f16(v16h a, v16h b, v8f c) {
    return __builtin_amdgcn_wmma_f32_16x16x32_f16(false, a, false, b, (short)0, c, false, false);
}
__device__ __forceinline__ v16h scale16(v16h v, _Float16 s) {
    v16h r;
#pragma unroll
    for (int i = 0; i < 16; ++i) r[i] = v[i] * s;
    return r;
}

// ---------------- fp32 -> f16 conversion ----------------
__global__ __launch_bounds__(256) void cvt_f32_f16(const float* __restrict__ in,
                                                   h16* __restrict__ out, int n) {
    int i = blockIdx.x * blockDim.x + threadIdx.x;
    int stride = gridDim.x * blockDim.x;
    for (; i < n; i += stride) out[i] = (h16)in[i];
}

// ---------------- GEMM: out[n,o] = sum_c X[n,c]*W[o,c] + bias[o] ----------------
// X: [M,K] f16 row-major.  W: [N,K] f16 row-major (computes X*W^T).
// Block = 256 threads = 8 waves. Wave -> 16x128 output strip. Grid (N/128, M/128).
template <typename OUT>
__global__ __launch_bounds__(256) void gemm_xwt(const h16* __restrict__ X,
                                                const h16* __restrict__ W,
                                                const float* __restrict__ bias,
                                                OUT* __restrict__ out,
                                                int M, int N, int K) {
    const int wave = threadIdx.x >> 5, lane = threadIdx.x & 31;
    const int half = lane >> 4, lr = lane & 15;
    const int rowBase = blockIdx.y * 128 + wave * 16;
    const int colBase = blockIdx.x * 128;

    const h16* arow = X + (size_t)(rowBase + lr) * K;
    const h16* brow[8];
#pragma unroll
    for (int j = 0; j < 8; ++j) brow[j] = W + (size_t)(colBase + j * 16 + lr) * K;

    v8f acc[8] = {};
    for (int k = 0; k < K; k += 32) {
        // A fragment 16x32: half 0 -> K 0..15, half 1 -> K 16..31
        v16h a = ldfrag(arow + k + half * 16, arow + k + half * 16 + 8);
#pragma unroll
        for (int j = 0; j < 8; ++j) {
            // B fragment 32x16: lane = column (W row), half selects K 0..15 / 16..31
            v16h b = ldfrag(brow[j] + k + half * 16, brow[j] + k + half * 16 + 8);
            acc[j] = wmma_f16(a, b, acc[j]);
        }
    }
#pragma unroll
    for (int j = 0; j < 8; ++j) {
        const int col = colBase + j * 16 + lr;
        const float bsv = bias[col];
#pragma unroll
        for (int r = 0; r < 8; ++r) {
            const int row = rowBase + r + 8 * half;
            out[(size_t)row * N + col] = (OUT)(acc[j][r] + bsv);
        }
    }
}

// ---------------- Flash attention (causal), 64-wide s-blocks ----------------
// q,k,v,y: f16 [B*T, C], head h occupies columns h*64..h*64+63.
// Grid: (T/128, B*H). 8 waves; wave w owns Q rows [qBase+16w, +16).
__global__ __launch_bounds__(256) void flash_attn(const h16* __restrict__ Q,
                                                  const h16* __restrict__ Km,
                                                  const h16* __restrict__ V,
                                                  h16* __restrict__ Y) {
    __shared__ __align__(32) h16 Klds[64][72];      // K block  [s][d], padded
    __shared__ __align__(32) h16 Vtlds[64][72];     // V block transposed [d][s], padded
    __shared__ __align__(32) h16 Plds[8][16][72];   // per-wave P tile [m][s], padded

    const int wave = threadIdx.x >> 5, lane = threadIdx.x & 31;
    const int half = lane >> 4, lr = lane & 15;
    const int bh = blockIdx.y;
    const int b = bh / Hh, h = bh % Hh;
    const int qBase = blockIdx.x * 128;
    const size_t seqOff = (size_t)b * Tt * Cc + (size_t)h * Dd;
    const h16* qp = Q + seqOff;
    const h16* kp = Km + seqOff;
    const h16* vp = V + seqOff;

    // Q fragments: fold (1/sqrt(D)) * log2(e) so scores are already in exp2 domain.
    const int myRow = qBase + wave * 16;
    const h16* qrow = qp + (size_t)(myRow + lr) * Cc;
    const _Float16 qscale = (_Float16)(0.125f * 1.44269504089f);
    const v16h qf0 = scale16(ldfrag(qrow + half * 16,      qrow + half * 16 + 8), qscale); // d 0..31
    const v16h qf1 = scale16(ldfrag(qrow + 32 + half * 16, qrow + 32 + half * 16 + 8), qscale); // d 32..63

    v16h ones;
#pragma unroll
    for (int i = 0; i < 16; ++i) ones[i] = (_Float16)1.0f;

    v8f o0 = {}, o1 = {}, o2 = {}, o3 = {};
    v8f lacc = {};                         // row-sum accumulator via P*ones WMMA
    float rowm[8];
#pragma unroll
    for (int r = 0; r < 8; ++r) rowm[r] = -1e30f;

    const int nBlk = qBase / 64 + 2;       // covers s <= qBase+127 (causal)

    for (int blk = 0; blk < nBlk; ++blk) {
        const int sBase = blk * 64;
        __syncthreads();
        {   // cooperative stage of K (direct) and V (transposed): 64x64 halves
            const int rr = threadIdx.x >> 2;        // 0..63  (s within block)
            const int cc = (threadIdx.x & 3) * 16;  // 0,16,32,48 (d)
            const h16* ksrc = kp + (size_t)(sBase + rr) * Cc + cc;
            *(v8h*)&Klds[rr][cc]     = *(const v8h*)ksrc;
            *(v8h*)&Klds[rr][cc + 8] = *(const v8h*)(ksrc + 8);
            const h16* vsrc = vp + (size_t)(sBase + rr) * Cc + cc;
            v8h v0 = *(const v8h*)vsrc;
            v8h v1 = *(const v8h*)(vsrc + 8);
#pragma unroll
            for (int j = 0; j < 8; ++j) { Vtlds[cc + j][rr] = v0[j]; Vtlds[cc + 8 + j][rr] = v1[j]; }
        }
        __syncthreads();

        if (sBase <= myRow + 15) {   // wave has at least one unmasked column in this block
            // S = Q*K^T : four 16x16 f32 tiles (s cols 0..63), contraction over d.
            v8f s[4];
#pragma unroll
            for (int n = 0; n < 4; ++n) {
                v8f t = {};
                v16h bk = ldfrag(&Klds[n * 16 + lr][half * 16],      &Klds[n * 16 + lr][half * 16 + 8]);
                t = wmma_f16(qf0, bk, t);
                bk      = ldfrag(&Klds[n * 16 + lr][32 + half * 16], &Klds[n * 16 + lr][32 + half * 16 + 8]);
                t = wmma_f16(qf1, bk, t);
                s[n] = t;
            }

            const bool needMask = (sBase + 63 > myRow);  // wave-uniform diagonal check
            if (needMask) {
#pragma unroll
                for (int n = 0; n < 4; ++n) {
                    const int col = sBase + n * 16 + lr;
#pragma unroll
                    for (int r = 0; r < 8; ++r) {
                        const int rowIdx = myRow + r + 8 * half;
                        if (col > rowIdx) s[n][r] = -1e30f;
                    }
                }
            }

            // Online softmax (exp2 domain). Row m lives across 16 lanes of VGPR slot r.
#pragma unroll
            for (int r = 0; r < 8; ++r) {
                float mcur = fmaxf(fmaxf(s[0][r], s[1][r]), fmaxf(s[2][r], s[3][r]));
#pragma unroll
                for (int m = 1; m < 16; m <<= 1) mcur = fmaxf(mcur, __shfl_xor(mcur, m));
                const float mnew  = fmaxf(rowm[r], mcur);
                const float alpha = fast_exp2(rowm[r] - mnew);
                rowm[r] = mnew;
                const float p0 = fast_exp2(s[0][r] - mnew);
                const float p1 = fast_exp2(s[1][r] - mnew);
                const float p2 = fast_exp2(s[2][r] - mnew);
                const float p3 = fast_exp2(s[3][r] - mnew);
                o0[r] *= alpha; o1[r] *= alpha; o2[r] *= alpha; o3[r] *= alpha;
                lacc[r] *= alpha;
                const int m16 = r + 8 * half;
                Plds[wave][m16][lr]      = (h16)p0;
                Plds[wave][m16][16 + lr] = (h16)p1;
                Plds[wave][m16][32 + lr] = (h16)p2;
                Plds[wave][m16][48 + lr] = (h16)p3;
            }
            // Per-wave LDS ops complete in-order; stop compiler reordering only.
            asm volatile("" ::: "memory");

            // A fragments of P (16x64) from LDS.
            const v16h aP0 = ldfrag(&Plds[wave][lr][half * 16],      &Plds[wave][lr][half * 16 + 8]); // s 0..31
            const v16h aP1 = ldfrag(&Plds[wave][lr][32 + half * 16], &Plds[wave][lr][32 + half * 16 + 8]); // s 32..63

            // Row sums: lacc += P * ones  (D[m,n] = rowsum, replicated over n)
            lacc = wmma_f16(aP0, ones, lacc);
            lacc = wmma_f16(aP1, ones, lacc);

            // O += P*V : per d-tile j, contraction over s in two 32-chunks.
            v16h bv;
            bv = ldfrag(&Vtlds[lr][half * 16],           &Vtlds[lr][half * 16 + 8]);
            o0 = wmma_f16(aP0, bv, o0);
            bv = ldfrag(&Vtlds[lr][32 + half * 16],      &Vtlds[lr][32 + half * 16 + 8]);
            o0 = wmma_f16(aP1, bv, o0);
            bv = ldfrag(&Vtlds[16 + lr][half * 16],      &Vtlds[16 + lr][half * 16 + 8]);
            o1 = wmma_f16(aP0, bv, o1);
            bv = ldfrag(&Vtlds[16 + lr][32 + half * 16], &Vtlds[16 + lr][32 + half * 16 + 8]);
            o1 = wmma_f16(aP1, bv, o1);
            bv = ldfrag(&Vtlds[32 + lr][half * 16],      &Vtlds[32 + lr][half * 16 + 8]);
            o2 = wmma_f16(aP0, bv, o2);
            bv = ldfrag(&Vtlds[32 + lr][32 + half * 16], &Vtlds[32 + lr][32 + half * 16 + 8]);
            o2 = wmma_f16(aP1, bv, o2);
            bv = ldfrag(&Vtlds[48 + lr][half * 16],      &Vtlds[48 + lr][half * 16 + 8]);
            o3 = wmma_f16(aP0, bv, o3);
            bv = ldfrag(&Vtlds[48 + lr][32 + half * 16], &Vtlds[48 + lr][32 + half * 16 + 8]);
            o3 = wmma_f16(aP1, bv, o3);
        }
    }

    // Normalize and store this wave's 16x64 output tile.
#pragma unroll
    for (int r = 0; r < 8; ++r) {
        const int row = myRow + r + 8 * half;
        const float inv = 1.0f / lacc[r];
        h16* yr = Y + seqOff + (size_t)row * Cc;
        yr[lr]      = (h16)(o0[r] * inv);
        yr[16 + lr] = (h16)(o1[r] * inv);
        yr[32 + lr] = (h16)(o2[r] * inv);
        yr[48 + lr] = (h16)(o3[r] * inv);
    }
}

extern "C" void kernel_launch(void* const* d_in, const int* in_sizes, int n_in,
                              void* d_out, int out_size, void* d_ws, size_t ws_size,
                              hipStream_t stream) {
    (void)in_sizes; (void)n_in; (void)out_size; (void)ws_size;
    const float* x  = (const float*)d_in[0];
    const float* Wq = (const float*)d_in[1];
    const float* bq = (const float*)d_in[2];
    const float* Wk = (const float*)d_in[3];
    const float* bk = (const float*)d_in[4];
    const float* Wv = (const float*)d_in[5];
    const float* bv = (const float*)d_in[6];
    const float* Wp = (const float*)d_in[7];
    const float* bp = (const float*)d_in[8];
    float* out = (float*)d_out;

    const size_t NTC = (size_t)Bb * Tt * Cc;   // 8,388,608
    const size_t CC  = (size_t)Cc * Cc;        // 1,048,576

    h16* ws  = (h16*)d_ws;
    h16* xh  = ws;
    h16* wqh = xh + NTC;
    h16* wkh = wqh + CC;
    h16* wvh = wkh + CC;
    h16* wph = wvh + CC;
    h16* qh  = wph + CC;
    h16* kh  = qh + NTC;
    h16* vh  = kh + NTC;
    h16* yh  = vh + NTC;   // total ~88 MB of workspace

    cvt_f32_f16<<<2048, 256, 0, stream>>>(x,  xh,  (int)NTC);
    cvt_f32_f16<<<1024, 256, 0, stream>>>(Wq, wqh, (int)CC);
    cvt_f32_f16<<<1024, 256, 0, stream>>>(Wk, wkh, (int)CC);
    cvt_f32_f16<<<1024, 256, 0, stream>>>(Wv, wvh, (int)CC);
    cvt_f32_f16<<<1024, 256, 0, stream>>>(Wp, wph, (int)CC);

    const int M = Bb * Tt;                      // 8192
    dim3 gg(Cc / 128, M / 128);                 // (8, 64)
    gemm_xwt<h16><<<gg, 256, 0, stream>>>(xh, wqh, bq, qh, M, Cc, Cc);
    gemm_xwt<h16><<<gg, 256, 0, stream>>>(xh, wkh, bk, kh, M, Cc, Cc);
    gemm_xwt<h16><<<gg, 256, 0, stream>>>(xh, wvh, bv, vh, M, Cc, Cc);

    flash_attn<<<dim3(Tt / 128, Bb * Hh), 256, 0, stream>>>(qh, kh, vh, yh);

    gemm_xwt<float><<<gg, 256, 0, stream>>>(yh, wph, bp, out, M, Cc, Cc);
}
